// LinearCrossEntropy_58325655880153
// MI455X (gfx1250) — matrix operand order repaired and checked by reading, not compiled, not run on hardware
//
#include <hip/hip_runtime.h>
#include <hip/hip_bf16.h>
#include <math.h>

// ---------------------------------------------------------------------------
// Fused linear + cross-entropy for MI455X (gfx1250, wave32).
//
//   logits[n,v] = dot(e[n,:], c[v,:]) + bias[v]
//   loss = mean_n( logsumexp_v(logits[n,:]) - logits[n, t_n] ),  t_n != -100
//
// Memory-bound on streaming c (823 MB fp32 / 412 MB bf16) once per 32-row
// stripe (256 stripes).  e-tile (32x4096) lives in LDS as bf16, transposed to
// [k][row] so it reads directly as WMMA B-fragments (lane=N=e-row, 16
// strided K values per lane).  c rows read directly as WMMA A-fragments (lane=M=vocab
// row, two contiguous 8-element K runs).  Each wave processes TWO adjacent
// vocab tiles per iteration: B-fragments are shared, giving 4 independent
// WMMA accumulation chains per wave (hides XDL RAW hazards) and a
// 4 global-b128 + 32 LDS bf16 reads + 4 WMMA steady-state per K-step.
// Bias (or -inf for the V tail) initializes the WMMA accumulators; a fully
// out-of-range tile keeps its accumulators at -inf and the online-softmax
// update becomes a no-op, so no tail branches in the hot loop.
// ---------------------------------------------------------------------------

typedef __bf16 bf16_t;
typedef bf16_t v16bf __attribute__((ext_vector_type(16)));
typedef bf16_t v8bf  __attribute__((ext_vector_type(8)));
typedef bf16_t v4bf  __attribute__((ext_vector_type(4)));
typedef float  v8f   __attribute__((ext_vector_type(8)));

#define IGNORE_INDEX (-100)
#define WAVES_PER_BLOCK 8
#define MROWS 32   // e-rows per workgroup

// ---------------- tiny utility kernels ----------------

__global__ void zero_ws_kernel(float* wsSum, int* wsCnt) {
  *wsSum = 0.0f;
  *wsCnt = 0;
}

__global__ void finalize_kernel(const float* wsSum, const int* wsCnt, float* out) {
  int c = *wsCnt;
  out[0] = *wsSum / (float)(c > 0 ? c : 1);
}

__global__ __launch_bounds__(256) void convert_c_kernel(const float* __restrict__ c,
                                                        bf16_t* __restrict__ cb,
                                                        long long n4) {
  long long i = (long long)blockIdx.x * blockDim.x + threadIdx.x;
  if (i >= n4) return;
  float4 f = ((const float4*)c)[i];
  v4bf o;
  o[0] = (bf16_t)f.x; o[1] = (bf16_t)f.y; o[2] = (bf16_t)f.z; o[3] = (bf16_t)f.w;
  ((v4bf*)cb)[i] = o;
}

// ---------------- device helpers ----------------

__device__ __forceinline__ void online_update(const v8f& acc, float& m, float& s,
                                              float& g, int tgt, int vb, int hi) {
  float tmax = acc[0];
#pragma unroll
  for (int i = 1; i < 8; ++i) tmax = fmaxf(tmax, acc[i]);
  float nm = fmaxf(m, tmax);
  if (nm > -INFINITY) {           // guard: all-(-inf) tail tile would give NaN
    float ssum = 0.0f;
#pragma unroll
    for (int i = 0; i < 8; ++i) ssum += __expf(acc[i] - nm);
    s = s * __expf(m - nm) + ssum;
    m = nm;
  }
#pragma unroll
  for (int i = 0; i < 8; ++i) {
    const int v = vb + hi * 8 + i;   // this lane's vocab index for acc[i]
    if (v == tgt) g += acc[i];
  }
}

__device__ __forceinline__ void pair_merge(float& m, float& s, float& g) {
  // lanes l and l+16 track the same e-row over disjoint vocab halves
  float om = __shfl_xor(m, 16);
  float os = __shfl_xor(s, 16);
  float og = __shfl_xor(g, 16);
  float M = fmaxf(m, om);
  float S = 0.0f;
  if (M > -INFINITY) S = s * __expf(m - M) + os * __expf(om - M);
  m = M; s = S; g = g + og;
}

__device__ __forceinline__ v8f bias_init(const float* __restrict__ bias,
                                         int vb, int hi, int V) {
  v8f bv;
#pragma unroll
  for (int i = 0; i < 8; ++i) {
    const int v = vb + hi * 8 + i;
    const float b = bias[min(v, V - 1)];
    bv[i] = (v < V) ? b : -INFINITY;
  }
  return bv;
}

// A-fragment loaders: two contiguous 8-element K runs from one c row.
__device__ __forceinline__ v16bf load_a_bf16(const bf16_t* __restrict__ cb,
                                             size_t rbase, int koff) {
  v8bf x0 = *(const v8bf*)(cb + rbase + koff);
  v8bf x1 = *(const v8bf*)(cb + rbase + koff + 16);
  return __builtin_shufflevector(x0, x1, 0, 1, 2, 3, 4, 5, 6, 7,
                                 8, 9, 10, 11, 12, 13, 14, 15);
}

__device__ __forceinline__ v16bf load_a_f32(const float* __restrict__ c,
                                            size_t rbase, int koff) {
  const float* cp = c + rbase + koff;
  float4 f0 = *(const float4*)(cp);
  float4 f1 = *(const float4*)(cp + 4);
  float4 f2 = *(const float4*)(cp + 16);
  float4 f3 = *(const float4*)(cp + 20);
  v16bf a;
  a[0]  = (bf16_t)f0.x; a[1]  = (bf16_t)f0.y; a[2]  = (bf16_t)f0.z; a[3]  = (bf16_t)f0.w;
  a[4]  = (bf16_t)f1.x; a[5]  = (bf16_t)f1.y; a[6]  = (bf16_t)f1.z; a[7]  = (bf16_t)f1.w;
  a[8]  = (bf16_t)f2.x; a[9]  = (bf16_t)f2.y; a[10] = (bf16_t)f2.z; a[11] = (bf16_t)f2.w;
  a[12] = (bf16_t)f3.x; a[13] = (bf16_t)f3.y; a[14] = (bf16_t)f3.z; a[15] = (bf16_t)f3.w;
  return a;
}

// ---------------- main fused kernel ----------------

template <bool USE_CB>
__global__ __launch_bounds__(256) void fused_lce_kernel(
    const float* __restrict__ e, const float* __restrict__ c,
    const bf16_t* __restrict__ cb, const float* __restrict__ bias,
    const int* __restrict__ targets, int N, int D, int V,
    float* wsSum, int* wsCnt)
{
  extern __shared__ __align__(64) char smem[];
  bf16_t* eT   = (bf16_t*)smem;                                   // [D][32] bf16
  float*  mrgM = (float*)(smem + (size_t)D * MROWS * sizeof(bf16_t));
  float*  mrgS = mrgM + WAVES_PER_BLOCK * MROWS;
  float*  mrgG = mrgS + WAVES_PER_BLOCK * MROWS;

  const int tid  = threadIdx.x;
  const int lane = tid & 31;
  const int wave = tid >> 5;
  const int hi   = lane >> 4;   // 0: lanes 0-15, 1: lanes 16-31
  const int l16  = lane & 15;
  const int nb   = blockIdx.x * MROWS;

  // ---- stage e tile into LDS: fp32 -> bf16, transposed to [k][row] ----
  {
    const int r    = tid >> 3;          // 0..31  (row within tile)
    const int k0   = (tid & 7) * 4;     // 0..28
    const int row  = min(nb + r, N - 1);
    const float* erow = e + (size_t)row * D;
    for (int k = k0; k < D; k += 32) {
      float4 f = *(const float4*)(erow + k);
      eT[(k + 0) * MROWS + r] = (bf16_t)f.x;
      eT[(k + 1) * MROWS + r] = (bf16_t)f.y;
      eT[(k + 2) * MROWS + r] = (bf16_t)f.z;
      eT[(k + 3) * MROWS + r] = (bf16_t)f.w;
    }
  }
  __syncthreads();

  const int tgt0 = targets[min(nb + l16, N - 1)];        // e-rows 0..15
  const int tgt1 = targets[min(nb + 16 + l16, N - 1)];   // e-rows 16..31

  // online-softmax state: [tile-pair half x][e-row half]
  float m00 = -INFINITY, s00 = 0.0f, g00 = 0.0f;   // tileA x rows 0..15
  float m01 = -INFINITY, s01 = 0.0f, g01 = 0.0f;   // tileA x rows 16..31
  float m10 = -INFINITY, s10 = 0.0f, g10 = 0.0f;   // tileB x rows 0..15
  float m11 = -INFINITY, s11 = 0.0f, g11 = 0.0f;   // tileB x rows 16..31

  const int numTiles = (V + 15) >> 4;
  const int numPairs = (numTiles + 1) >> 1;
  for (int p = wave; p < numPairs; p += WAVES_PER_BLOCK) {
    const int vb0 = (p << 5);        // tileA vocab base
    const int vb1 = vb0 + 16;        // tileB vocab base (may be fully OOB)

    v8f acc00 = bias_init(bias, vb0, hi, V);
    v8f acc01 = acc00;
    v8f acc10 = bias_init(bias, vb1, hi, V);
    v8f acc11 = acc10;

    // A sources: lanes supply vocab rows vb0+l16 and vb1+l16 (clamped; garbage
    // products land on -inf accumulators and vanish).
    const size_t rbase0 = (size_t)min(vb0 + l16, V - 1) * D;
    const size_t rbase1 = (size_t)min(vb1 + l16, V - 1) * D;

    for (int kb = 0; kb < D; kb += 32) {
      const int koff = kb + hi * 8;   // lanes<16: K 0..7/16..23, lanes>=16: 8..15/24..31
      v16bf a0, a1;
      if (USE_CB) {
        a0 = load_a_bf16(cb, rbase0, koff);
        a1 = load_a_bf16(cb, rbase1, koff);
        if (kb + 32 < D) {
          __builtin_prefetch(cb + rbase0 + koff + 32, 0, 3);
          __builtin_prefetch(cb + rbase1 + koff + 32, 0, 3);
        }
      } else {
        a0 = load_a_f32(c, rbase0, koff);
        a1 = load_a_f32(c, rbase1, koff);
        if (kb + 32 < D) {
          __builtin_prefetch(c + rbase0 + koff + 32, 0, 3);
          __builtin_prefetch(c + rbase1 + koff + 32, 0, 3);
        }
      }

      // B fragments from LDS eT: lane = e-row (l16), elements = same K runs as A.
      // Shared by both vocab tiles.
      v16bf b0, b1;
#pragma unroll
      for (int s = 0; s < 16; ++s) {
        const int k = koff + s + (s >= 8 ? 8 : 0);
        b0[s] = eT[(size_t)k * MROWS + l16];        // rows 0..15
        b1[s] = eT[(size_t)k * MROWS + 16 + l16];   // rows 16..31
      }

      // 4 independent accumulation chains -> XDL pipeline stays busy
      acc00 = __builtin_amdgcn_wmma_f32_16x16x32_bf16(false, a0, false, b0,
                                                      (short)0, acc00, false, false);
      acc01 = __builtin_amdgcn_wmma_f32_16x16x32_bf16(false, a0, false, b1,
                                                      (short)0, acc01, false, false);
      acc10 = __builtin_amdgcn_wmma_f32_16x16x32_bf16(false, a1, false, b0,
                                                      (short)0, acc10, false, false);
      acc11 = __builtin_amdgcn_wmma_f32_16x16x32_bf16(false, a1, false, b1,
                                                      (short)0, acc11, false, false);
    }

    online_update(acc00, m00, s00, g00, tgt0, vb0, hi);
    online_update(acc01, m01, s01, g01, tgt1, vb0, hi);
    online_update(acc10, m10, s10, g10, tgt0, vb1, hi);
    online_update(acc11, m11, s11, g11, tgt1, vb1, hi);
  }

  // fold tileB state into tileA state (same e-rows, disjoint vocab subsets)
  {
    float nm = fmaxf(m00, m10);
    if (nm > -INFINITY) s00 = s00 * __expf(m00 - nm) + s10 * __expf(m10 - nm);
    m00 = nm; g00 += g10;
    nm = fmaxf(m01, m11);
    if (nm > -INFINITY) s01 = s01 * __expf(m01 - nm) + s11 * __expf(m11 - nm);
    m01 = nm; g01 += g11;
  }

  // merge the two vocab halves (lane <-> lane+16)
  pair_merge(m00, s00, g00);
  pair_merge(m01, s01, g01);

  // per-wave partials -> LDS
  if (lane < 16) {
    mrgM[wave * MROWS + l16]      = m00;
    mrgS[wave * MROWS + l16]      = s00;
    mrgG[wave * MROWS + l16]      = g00;
    mrgM[wave * MROWS + 16 + l16] = m01;
    mrgS[wave * MROWS + 16 + l16] = s01;
    mrgG[wave * MROWS + 16 + l16] = g01;
  }
  __syncthreads();

  // wave 0 merges the 8 per-wave online-softmax states per row
  if (tid < 32) {
    float M = -INFINITY;
#pragma unroll
    for (int w = 0; w < WAVES_PER_BLOCK; ++w)
      M = fmaxf(M, mrgM[w * MROWS + tid]);
    float S = 0.0f, G = 0.0f;
#pragma unroll
    for (int w = 0; w < WAVES_PER_BLOCK; ++w) {
      const float mw = mrgM[w * MROWS + tid];
      S += mrgS[w * MROWS + tid] * __expf(mw - M);
      G += mrgG[w * MROWS + tid];
    }
    const int  row   = nb + tid;
    const int  tv    = targets[min(row, N - 1)];
    const bool valid = (row < N) && (tv != IGNORE_INDEX);
    float nll = valid ? (M + __logf(S) - G) : 0.0f;
    int   cnt = valid ? 1 : 0;
#pragma unroll
    for (int off = 16; off > 0; off >>= 1) {
      nll += __shfl_xor(nll, off);
      cnt += __shfl_xor(cnt, off);
    }
    if (tid == 0) {
      atomicAdd(wsSum, nll);
      atomicAdd(wsCnt, cnt);
    }
  }
}

// ---------------- launch ----------------

extern "C" void kernel_launch(void* const* d_in, const int* in_sizes, int n_in,
                              void* d_out, int out_size, void* d_ws, size_t ws_size,
                              hipStream_t stream) {
  const float* e       = (const float*)d_in[0];
  const float* c       = (const float*)d_in[1];
  const float* bias    = (const float*)d_in[2];
  const int*   targets = (const int*)d_in[3];

  const int N = in_sizes[3];             // 8192
  const int V = in_sizes[2];             // 50257
  const int D = in_sizes[0] / N;         // 4096

  float* wsSum = (float*)d_ws;
  int*   wsCnt = (int*)((char*)d_ws + sizeof(float));
  bf16_t* cb   = (bf16_t*)((char*)d_ws + 256);
  const size_t cbBytes = (size_t)V * (size_t)D * sizeof(bf16_t);
  const bool useCb = (ws_size >= 256 + cbBytes);

  const size_t smemBytes =
      (size_t)D * MROWS * sizeof(bf16_t) + 3u * WAVES_PER_BLOCK * MROWS * sizeof(float);

  (void)hipFuncSetAttribute(reinterpret_cast<const void*>(&fused_lce_kernel<true>),
                            hipFuncAttributeMaxDynamicSharedMemorySize, (int)smemBytes);
  (void)hipFuncSetAttribute(reinterpret_cast<const void*>(&fused_lce_kernel<false>),
                            hipFuncAttributeMaxDynamicSharedMemorySize, (int)smemBytes);

  zero_ws_kernel<<<1, 1, 0, stream>>>(wsSum, wsCnt);

  const int blocks = N / MROWS;          // 256 row stripes
  if (useCb) {
    const long long n4 = ((long long)V * (long long)D) / 4;
    const int cblocks = (int)((n4 + 255) / 256);
    convert_c_kernel<<<cblocks, 256, 0, stream>>>(c, cb, n4);
    fused_lce_kernel<true><<<blocks, 256, smemBytes, stream>>>(
        e, c, cb, bias, targets, N, D, V, wsSum, wsCnt);
  } else {
    fused_lce_kernel<false><<<blocks, 256, smemBytes, stream>>>(
        e, c, nullptr, bias, targets, N, D, V, wsSum, wsCnt);
  }

  finalize_kernel<<<1, 1, 0, stream>>>(wsSum, wsCnt, (float*)d_out);
}
